// LSTMAutoencoder_19464791785503
// MI455X (gfx1250) — compile-verified, matrix operand order checked by reading, not run on hardware
//
#include <hip/hip_runtime.h>
#include <cstdint>
#include <cstddef>

// LSTM autoencoder: B=128, T=512, F=64, H=512.
// Persistent-kernel design: 16 workgroups x 256 threads (8 wave32s).
// Whh/Wih (bf16) resident in LDS per-WGP in k-chunk-swizzled layout so every
// B-tile ds_load_b128 uses an immediate offset; h double-buffered bf16 in L2;
// one grid barrier per time step; v_wmma_f32_16x16x32_bf16 for all GEMMs.

#define B_   128
#define T_   512
#define F_   64
#define H_   512
#define NWG  16

// LDS weight layout: chunk kc holds 128 rows x 32 bf16 contiguously:
//   elem(kc, rloc, kk) = kc*4096 + rloc*32 + kk
// Whh: kc = 0..15 (K=512).  Wih: kc = 0..1 (K=64) at WIH_OFF.
#define KCHUNK_ELEMS 4096          // 128 rows * 32
#define WIH_OFF      (16 * KCHUNK_ELEMS)   // 65536 elems

typedef __attribute__((ext_vector_type(16))) __bf16 v16bf;
typedef __attribute__((ext_vector_type(8)))  float  v8f;

union ABU {
    v16bf v;
    uint4 q[2];
    __bf16 s[16];
};

__device__ __forceinline__ unsigned short f2bf(float f) {
    union { __bf16 b; unsigned short s; } u;
    u.b = (__bf16)f;                 // native f32->bf16 convert (RNE)
    return u.s;
}

__device__ __forceinline__ float fast_tanh(float x) {
#if __has_builtin(__builtin_amdgcn_tanhf)
    return __builtin_amdgcn_tanhf(x);   // v_tanh_f32 (TRANS)
#else
    return tanhf(x);
#endif
}

__device__ __forceinline__ float sigmoidf_(float x) {
    return __builtin_amdgcn_rcpf(1.0f + __expf(-x));
}

// A/B operand tile (16x32 bf16), lane layout per CDNA5 ISA 7.12.2:
// lane (0..15): row = lane, holds K = kbase..kbase+7 and kbase+16..kbase+23
// lane (16..31): same rows, kbase shifted by 8.  Caller passes p already at
// rowbase + kbase; the two quads sit at p and p+16 elements.
__device__ __forceinline__ v16bf ld_bf16_tile(const unsigned short* p) {
    ABU a;
    a.q[0] = *(const uint4*)(p);
    a.q[1] = *(const uint4*)(p + 16);
    return a.v;
}

__device__ __forceinline__ v16bf ld_f32_tile(const float* p) {
    ABU a;
    float4 v0 = *(const float4*)(p);
    float4 v1 = *(const float4*)(p + 4);
    float4 v2 = *(const float4*)(p + 16);
    float4 v3 = *(const float4*)(p + 20);
    a.s[0] = (__bf16)v0.x; a.s[1] = (__bf16)v0.y; a.s[2]  = (__bf16)v0.z; a.s[3]  = (__bf16)v0.w;
    a.s[4] = (__bf16)v1.x; a.s[5] = (__bf16)v1.y; a.s[6]  = (__bf16)v1.z; a.s[7]  = (__bf16)v1.w;
    a.s[8] = (__bf16)v2.x; a.s[9] = (__bf16)v2.y; a.s[10] = (__bf16)v2.z; a.s[11] = (__bf16)v2.w;
    a.s[12] = (__bf16)v3.x; a.s[13] = (__bf16)v3.y; a.s[14] = (__bf16)v3.z; a.s[15] = (__bf16)v3.w;
    return a.v;
}

// Load this WG's 128 gate rows of Whh (K=512) and Wih (K=64) into the
// k-chunk-swizzled LDS layout as bf16.
// LDS row rloc (0..127) <-> global gate row (rloc/32)*512 + wg*32 + (rloc%32).
__device__ __forceinline__ void load_weights_lds(unsigned short* ldsW,
                                                 const float* __restrict__ Whh,
                                                 const float* __restrict__ Wih,
                                                 int wg, int tid) {
    for (int i = tid; i < 128 * (H_ / 4); i += 256) {
        int rloc = i / (H_ / 4);
        int kq   = i % (H_ / 4);
        int k    = kq * 4;
        int grow = (rloc >> 5) * H_ + wg * 32 + (rloc & 31);
        float4 v = ((const float4*)(Whh + (size_t)grow * H_))[kq];
        unsigned short* d = ldsW + (k >> 5) * KCHUNK_ELEMS + rloc * 32 + (k & 31);
        d[0] = f2bf(v.x); d[1] = f2bf(v.y); d[2] = f2bf(v.z); d[3] = f2bf(v.w);
    }
    for (int i = tid; i < 128 * (F_ / 4); i += 256) {
        int rloc = i / (F_ / 4);
        int kq   = i % (F_ / 4);
        int k    = kq * 4;
        int grow = (rloc >> 5) * H_ + wg * 32 + (rloc & 31);
        float4 v = ((const float4*)(Wih + (size_t)grow * F_))[kq];
        unsigned short* d = ldsW + WIH_OFF + (k >> 5) * KCHUNK_ELEMS + rloc * 32 + (k & 31);
        d[0] = f2bf(v.x); d[1] = f2bf(v.y); d[2] = f2bf(v.z); d[3] = f2bf(v.w);
    }
}

__device__ __forceinline__ void grid_barrier(int* counters, int id) {
    __syncthreads();
    if (threadIdx.x == 0) {
        __hip_atomic_fetch_add(&counters[id], 1, __ATOMIC_RELEASE, __HIP_MEMORY_SCOPE_AGENT);
        while (__hip_atomic_load(&counters[id], __ATOMIC_ACQUIRE, __HIP_MEMORY_SCOPE_AGENT) < NWG) {
            __builtin_amdgcn_s_sleep(2);
        }
    }
    __syncthreads();
}

// One LSTM step for this wave: gates = h@Whh^T + x_t@Wih^T + bias, then cell
// update; writes h_new (bf16) to the write buffer; c stays in registers.
__device__ __forceinline__ void lstm_step(const unsigned short* __restrict__ hrd,
                                          unsigned short* __restrict__ hwr,
                                          const float* __restrict__ x, int t,
                                          const unsigned short* ldsW,
                                          const float bias_[4][2], float c[2][8],
                                          int wg, int m0, int lm, int l16) {
    v8f acc[4][2];
#pragma unroll
    for (int g = 0; g < 4; ++g)
#pragma unroll
        for (int nt = 0; nt < 2; ++nt) {
            v8f tv;
#pragma unroll
            for (int e = 0; e < 8; ++e) tv[e] = bias_[g][nt];
            acc[g][nt] = tv;
        }

    // Per-lane invariant component of the B addresses (row=lm, quad=l16).
    const unsigned short* bbase  = ldsW + lm * 32 + l16 * 8;
    const unsigned short* bbase2 = ldsW + WIH_OFF + lm * 32 + l16 * 8;
    const unsigned short* arow   = hrd + (size_t)(m0 + lm) * H_;

#pragma unroll 4
    for (int k0 = 0; k0 < H_; k0 += 32) {
        v16bf A = ld_bf16_tile(arow + k0 + l16 * 8);
        const unsigned short* bp = bbase + (k0 >> 5) * KCHUNK_ELEMS;
#pragma unroll
        for (int g = 0; g < 4; ++g)
#pragma unroll
            for (int nt = 0; nt < 2; ++nt) {
                v16bf Bm = ld_bf16_tile(bp + (g * 32 + nt * 16) * 32);
                acc[g][nt] = __builtin_amdgcn_wmma_f32_16x16x32_bf16(
                    false, A, false, Bm, (short)0, acc[g][nt], false, false);
            }
    }

    // x_t contribution (K = F = 64)
    const float* xrow = x + ((size_t)(m0 + lm) * T_ + t) * F_;
#pragma unroll
    for (int k0 = 0; k0 < F_; k0 += 32) {
        v16bf A = ld_f32_tile(xrow + k0 + l16 * 8);
        const unsigned short* bp = bbase2 + (k0 >> 5) * KCHUNK_ELEMS;
#pragma unroll
        for (int g = 0; g < 4; ++g)
#pragma unroll
            for (int nt = 0; nt < 2; ++nt) {
                v16bf Bm = ld_bf16_tile(bp + (g * 32 + nt * 16) * 32);
                acc[g][nt] = __builtin_amdgcn_wmma_f32_16x16x32_bf16(
                    false, A, false, Bm, (short)0, acc[g][nt], false, false);
            }
    }

    // cell update; C/D layout: lanes 0-15 -> M = e, lanes 16-31 -> M = e+8
#pragma unroll
    for (int nt = 0; nt < 2; ++nt) {
        int col = wg * 32 + nt * 16 + lm;
#pragma unroll
        for (int e = 0; e < 8; ++e) {
            float iv = acc[0][nt][e];
            float fv = acc[1][nt][e];
            float gv = acc[2][nt][e];
            float ov = acc[3][nt][e];
            float cn = sigmoidf_(fv) * c[nt][e] + sigmoidf_(iv) * fast_tanh(gv);
            c[nt][e] = cn;
            float hn = sigmoidf_(ov) * fast_tanh(cn);
            int row = m0 + e + l16 * 8;
            hwr[(size_t)row * H_ + col] = f2bf(hn);
        }
    }
}

__global__ void init_kernel(const float* __restrict__ Wout,
                            unsigned short* __restrict__ woutb,
                            int* __restrict__ counters,
                            unsigned short* __restrict__ hbuf) {
    int i = blockIdx.x * blockDim.x + threadIdx.x;   // 131072 threads
    if (i < 2048) counters[i] = 0;
    if (i < 2 * B_ * H_) hbuf[i] = 0;                // zero both h buffers
    if (i < F_ * H_) woutb[i] = f2bf(Wout[i]);
}

__launch_bounds__(256, 1)
__global__ void lstm_persistent(const float* __restrict__ x,
                                const float* __restrict__ eWih, const float* __restrict__ eWhh,
                                const float* __restrict__ eBih, const float* __restrict__ eBhh,
                                const float* __restrict__ dWih, const float* __restrict__ dWhh,
                                const float* __restrict__ dBih, const float* __restrict__ dBhh,
                                const float* __restrict__ bout,
                                const unsigned short* __restrict__ woutb,
                                unsigned short* __restrict__ hbuf,
                                int* __restrict__ counters,
                                float* __restrict__ out) {
    __shared__ __align__(16) unsigned short ldsW[128 * H_ + 128 * F_];  // 144 KB

    const int wg   = blockIdx.x;
    const int tid  = threadIdx.x;
    const int wv   = tid >> 5;
    const int lane = tid & 31;
    const int lm   = lane & 15;
    const int l16  = (lane >> 4) & 1;
    const int m0   = wv * 16;

    // ---------------- encoder ----------------
    load_weights_lds(ldsW, eWhh, eWih, wg, tid);
    __syncthreads();

    float bias_[4][2];
#pragma unroll
    for (int g = 0; g < 4; ++g)
#pragma unroll
        for (int nt = 0; nt < 2; ++nt) {
            int col = g * H_ + wg * 32 + nt * 16 + lm;
            bias_[g][nt] = eBih[col] + eBhh[col];
        }

    float c[2][8];
#pragma unroll
    for (int nt = 0; nt < 2; ++nt)
#pragma unroll
        for (int e = 0; e < 8; ++e) c[nt][e] = 0.0f;

    int bar = 0;
    for (int t = 0; t < T_; ++t) {
        int u = t;
        const unsigned short* hrd = hbuf + (size_t)(u & 1) * B_ * H_;
        unsigned short*       hwr = hbuf + (size_t)((u + 1) & 1) * B_ * H_;
        lstm_step(hrd, hwr, x, t, ldsW, bias_, c, wg, m0, lm, l16);
        grid_barrier(counters, bar++);
    }

    // c_enc -> second output region
#pragma unroll
    for (int nt = 0; nt < 2; ++nt) {
        int col = wg * 32 + nt * 16 + lm;
#pragma unroll
        for (int e = 0; e < 8; ++e) {
            int row = m0 + e + l16 * 8;
            out[(size_t)B_ * T_ * F_ + (size_t)row * H_ + col] = c[nt][e];
        }
    }

    // ---------------- decoder ----------------
    load_weights_lds(ldsW, dWhh, dWih, wg, tid);
    __syncthreads();

#pragma unroll
    for (int g = 0; g < 4; ++g)
#pragma unroll
        for (int nt = 0; nt < 2; ++nt) {
            int col = g * H_ + wg * 32 + nt * 16 + lm;
            bias_[g][nt] = dBih[col] + dBhh[col];
        }

    // projection tile for waves 0,1 of each WG: 32 tiles cover 128x64
    const int ptile = wg * 2 + wv;
    const int m0p   = (ptile >> 2) * 16;
    const int n0p   = (ptile & 3) * 16;
    float boutv = 0.0f;
    if (wv < 2) boutv = bout[n0p + lm];

    for (int s = 0; s < T_; ++s) {
        int u  = T_ + s;
        int tx = T_ - 1 - s;   // reversed input; also the output time index
        const unsigned short* hrd = hbuf + (size_t)(u & 1) * B_ * H_;
        unsigned short*       hwr = hbuf + (size_t)((u + 1) & 1) * B_ * H_;

        // out[:, tx] = h @ Wout^T + bout, using pre-update h
        if (wv < 2) {
            v8f accp;
#pragma unroll
            for (int e = 0; e < 8; ++e) accp[e] = boutv;
            const unsigned short* arow = hrd   + (size_t)(m0p + lm) * H_;
            const unsigned short* brow = woutb + (size_t)(n0p + lm) * H_;
#pragma unroll 4
            for (int k0 = 0; k0 < H_; k0 += 32) {
                v16bf A  = ld_bf16_tile(arow + k0 + l16 * 8);
                v16bf Bm = ld_bf16_tile(brow + k0 + l16 * 8);
                accp = __builtin_amdgcn_wmma_f32_16x16x32_bf16(
                    false, A, false, Bm, (short)0, accp, false, false);
            }
#pragma unroll
            for (int e = 0; e < 8; ++e) {
                int row = m0p + e + l16 * 8;
                out[(size_t)row * T_ * F_ + (size_t)tx * F_ + n0p + lm] = accp[e];
            }
        }

        lstm_step(hrd, hwr, x, tx, ldsW, bias_, c, wg, m0, lm, l16);
        grid_barrier(counters, bar++);
    }
}

extern "C" void kernel_launch(void* const* d_in, const int* in_sizes, int n_in,
                              void* d_out, int out_size, void* d_ws, size_t ws_size,
                              hipStream_t stream) {
    (void)in_sizes; (void)n_in; (void)out_size; (void)ws_size;
    const float* x    = (const float*)d_in[0];
    const float* eWih = (const float*)d_in[1];
    const float* eWhh = (const float*)d_in[2];
    const float* eBih = (const float*)d_in[3];
    const float* eBhh = (const float*)d_in[4];
    const float* dWih = (const float*)d_in[5];
    const float* dWhh = (const float*)d_in[6];
    const float* dBih = (const float*)d_in[7];
    const float* dBhh = (const float*)d_in[8];
    const float* Wout = (const float*)d_in[9];
    const float* bout = (const float*)d_in[10];
    float* out = (float*)d_out;

    // workspace layout: [0, 8KB) barrier counters | [8KB, 264KB) h double buffer
    // (bf16) | [264KB, 328KB) Wout bf16
    int*            counters = (int*)d_ws;
    unsigned short* hbuf     = (unsigned short*)((char*)d_ws + 8192);
    unsigned short* woutb    = (unsigned short*)((char*)d_ws + 8192 + 2 * 2 * B_ * H_);

    init_kernel<<<512, 256, 0, stream>>>(Wout, woutb, counters, hbuf);
    lstm_persistent<<<NWG, 256, 0, stream>>>(x, eWih, eWhh, eBih, eBhh,
                                             dWih, dWhh, dBih, dBhh,
                                             bout, woutb, hbuf, counters, out);
}
